// UNET_LSTM_BIDIRECTIONAL_AUTOENCODER_TACA_4_76235669504212
// MI455X (gfx1250) — matrix operand order repaired and checked by reading, not compiled, not run on hardware
//
#include <hip/hip_runtime.h>

// ---------------------------------------------------------------------------
// UNet + BiLSTM autoencoder forward for MI455X (gfx1250, wave32, WMMA).
// All GEMM-shaped work (3x3 convs as implicit GEMM, convT2 as 4x 1x1 GEMM,
// LSTM input/recurrent projections) runs through v_wmma_f32_16x16x32_f16
// with fp32 accumulation. Activations are channel-last f16; weights are
// pre-packed f16. GEMM staging uses gfx1250 async global->LDS b128 copies.
// ---------------------------------------------------------------------------

#ifndef USE_ASYNC_LDS
#define USE_ASYNC_LDS 1
#endif

typedef _Float16 f16;
typedef __attribute__((ext_vector_type(16))) _Float16 v16h;
typedef __attribute__((ext_vector_type(8)))  float    v8f;

#define DEV __device__ __forceinline__

static constexpr int Bsz = 16, Tn = 24, Cch = 8, Pn = 64;
static constexpr int NIMG = Bsz * Tn;          // 384
static constexpr long OUT_CODE = 0;
static constexpr long OUT_IMG  = 4096;                                  // B*256
static constexpr long OUT_AW   = OUT_IMG + (long)NIMG * Cch * Pn * Pn;  // +12582912
static constexpr long OUT_CA   = OUT_AW + Bsz * Tn;

DEV float sigf(float x) { return 1.0f / (1.0f + __expf(-x)); }

DEV float wave_reduce_sum(float v) {
  #pragma unroll
  for (int off = 16; off > 0; off >>= 1) v += __shfl_xor(v, off, 32);
  return v;
}

DEV int4 sel4(int4 v, bool m) {
  int4 z;
  z.x = m ? v.x : 0; z.y = m ? v.y : 0; z.z = m ? v.z : 0; z.w = m ? v.w : 0;
  return z;
}

// One 16x16x32 WMMA step from LDS staging with row stride `rh` halves.
// As: [16][rh] f16 (row m, K in cols 0..31 of this chunk).
// Bs: [16][rh] f16 (row n, K in cols).
DEV void wmma_acc_s(v8f& acc, const f16* As, const f16* Bs, int lane, int rh) {
  const int half = lane >> 4;
  const int q    = lane & 15;
  const f16* ar = As + q * rh;
  const f16* br = Bs + q * rh;
  v16h a, b;
  #pragma unroll
  for (int e = 0; e < 16; ++e) {
    const int g = e >> 3, r = e & 7;
    a[e] = ar[r + 8 * ((g << 1) + half)];   // A: K = r + 8*(2g + half)
    b[e] = br[r + 8 * ((g << 1) + half)];   // B: K = r + 8*(2g + half)
  }
  acc = __builtin_amdgcn_wmma_f32_16x16x32_f16(
      false, a, false, b, (short)0, acc, false, false);
}

// ---------------------------------------------------------------------------
// GEMM: C[M,N] = A[M,K] * B[N,K]^T (+bias, optional relu).
// Requires M%16==0, N%16==0, K%64==0 (true for all call sites).
// One wave per 16x16 tile; K stepped by 64 -> 2 WMMAs per staged tile.
// Staging: async global->LDS b128 (ASYNCcnt) or int4 via VGPRs.
// ---------------------------------------------------------------------------
__global__ void __launch_bounds__(128)
k_gemm_nt(const f16* __restrict__ A, const f16* __restrict__ B,
          const float* __restrict__ bias, float* __restrict__ Cf,
          f16* __restrict__ Ch, int M, int N, int K, int relu, int tiles) {
  __shared__ __align__(16) f16 lds[4][2][1024];   // per wave: A[16][64] + B[16][64]
  const int lane = threadIdx.x & 31, wave = threadIdx.x >> 5;
  const int tile = blockIdx.x * 4 + wave;
  if (tile >= tiles) return;
  const int nTiles = N >> 4;
  const int nt = tile % nTiles, mt = tile / nTiles;
  const int mBase = mt << 4, nBase = nt << 4;
  const int r = lane & 15, seg = lane >> 4;      // lane stages 64B of row r
  const f16* pa = A + (long)(mBase + r) * K + seg * 32;
  const f16* pb = B + (long)(nBase + r) * K + seg * 32;
  f16* As = lds[wave][0];
  f16* Bs = lds[wave][1];
#if USE_ASYNC_LDS
  const unsigned ldsA = (unsigned)(wave * 4096 + r * 128 + seg * 64);
  const unsigned ldsB = ldsA + 2048u;
#endif
  v8f acc = {};
  for (int k0 = 0; k0 < K; k0 += 64) {
    if (k0 + 64 < K) __builtin_prefetch(pa + 64, 0, 3);
#if USE_ASYNC_LDS
    const unsigned long long ga = (unsigned long long)pa;
    const unsigned long long gb = (unsigned long long)pb;
    asm volatile(
        "global_load_async_to_lds_b128 %0, %2, off\n\t"
        "global_load_async_to_lds_b128 %0, %2, off offset:16\n\t"
        "global_load_async_to_lds_b128 %0, %2, off offset:32\n\t"
        "global_load_async_to_lds_b128 %0, %2, off offset:48\n\t"
        "global_load_async_to_lds_b128 %1, %3, off\n\t"
        "global_load_async_to_lds_b128 %1, %3, off offset:16\n\t"
        "global_load_async_to_lds_b128 %1, %3, off offset:32\n\t"
        "global_load_async_to_lds_b128 %1, %3, off offset:48"
        :: "v"(ldsA), "v"(ldsB), "v"(ga), "v"(gb) : "memory");
    asm volatile("s_wait_asynccnt 0" ::: "memory");
#else
    const int4* qa = (const int4*)pa;
    const int4* qb = (const int4*)pb;
    int4* da = (int4*)(As + r * 64 + seg * 32);
    int4* db = (int4*)(Bs + r * 64 + seg * 32);
    #pragma unroll
    for (int i = 0; i < 4; ++i) { da[i] = qa[i]; db[i] = qb[i]; }
#endif
    __builtin_amdgcn_wave_barrier();
    wmma_acc_s(acc, As,      Bs,      lane, 64);
    wmma_acc_s(acc, As + 32, Bs + 32, lane, 64);
    __builtin_amdgcn_wave_barrier();
    pa += 64;
    pb += 64;
  }
  const int n = nBase + (lane & 15);
  const int hl = lane >> 4;
  const float bv = bias ? bias[n] : 0.f;
  #pragma unroll
  for (int j = 0; j < 8; ++j) {
    const int m = mBase + j + (hl << 3);
    float v = acc[j] + bv;
    if (relu && v < 0.f) v = 0.f;
    if (Cf) Cf[(long)m * N + n] = v;
    if (Ch) Ch[(long)m * N + n] = (f16)v;
  }
}

// ---------------------------------------------------------------------------
// 3x3 SAME conv as implicit GEMM. in: [nImg][H][W][Cin] f16 (channel-last).
// wp: [9][Cout][Cin] f16. out: [nImg][H][W][Cout] f16. 16 px x 16 cout / wave.
// HW%16==0 and Cout%16==0 at all call sites; border taps and Cin<32 are
// handled branchlessly (clamped address + component select to zero).
// ---------------------------------------------------------------------------
__global__ void __launch_bounds__(128)
k_conv3x3(const f16* __restrict__ in, const f16* __restrict__ wp,
          const float* __restrict__ bias, f16* __restrict__ out,
          int nImg, int H, int W, int Cin, int Cout, int relu, int tiles) {
  __shared__ __align__(16) f16 lds[4][2][512];
  const int lane = threadIdx.x & 31, wave = threadIdx.x >> 5;
  const int tile = blockIdx.x * 4 + wave;
  if (tile >= tiles) return;
  const int HW = H * W;
  const int mTiles = HW >> 4;
  const int nTiles = Cout >> 4;
  const int nt = tile % nTiles;
  const int t2 = tile / nTiles;
  const int mt = t2 % mTiles;
  const int img = t2 / mTiles;
  const int mBase = mt << 4, nBase = nt << 4;
  f16* As = lds[wave][0];
  f16* Bs = lds[wave][1];
  const int r = lane & 15, seg = lane >> 4;
  const int p = mBase + r;
  const int y = p / W, x = p % W;
  const int bRow = nBase + r;
  v8f acc = {};
  const int ckN = (Cin + 31) >> 5;
  for (int tap = 0; tap < 9; ++tap) {
    const int dy = tap / 3 - 1, dx = tap % 3 - 1;
    const int ys = y + dy, xs = x + dx;
    const bool vrow = ((unsigned)ys < (unsigned)H) && ((unsigned)xs < (unsigned)W);
    const int ysc = vrow ? ys : 0, xsc = vrow ? xs : 0;
    const f16* src  = in + (((long)img * H + ysc) * W + xsc) * Cin;
    const f16* wrow = wp + ((long)tap * Cout + bRow) * Cin;
    for (int ck = 0; ck < ckN; ++ck) {
      const int kb = (ck << 5) + (seg << 4);           // halves
      const bool cv0 = (kb + 8)  <= Cin;               // 16B block validity
      const bool cv1 = (kb + 16) <= Cin;
      int4 a0 = sel4(*(const int4*)(src + kb),      vrow && cv0);
      int4 a1 = sel4(*(const int4*)(src + kb + 8),  vrow && cv1);
      int4 b0 = sel4(*(const int4*)(wrow + kb),     cv0);
      int4 b1 = sel4(*(const int4*)(wrow + kb + 8), cv1);
      int4* da = (int4*)(As + r * 32 + (seg << 4));
      int4* db = (int4*)(Bs + r * 32 + (seg << 4));
      da[0] = a0; da[1] = a1;
      db[0] = b0; db[1] = b1;
      __builtin_amdgcn_wave_barrier();
      wmma_acc_s(acc, As, Bs, lane, 32);
      __builtin_amdgcn_wave_barrier();
    }
  }
  const int n = nBase + (lane & 15);
  const int hl = lane >> 4;
  const float bv = bias[n];
  #pragma unroll
  for (int j = 0; j < 8; ++j) {
    const int m = mBase + j + (hl << 3);
    float v = acc[j] + bv;
    if (relu && v < 0.f) v = 0.f;
    out[((long)img * HW + m) * Cout + n] = (f16)v;
  }
}

// ---------------------------------------------------------------------------
// ConvTranspose2d(k=2,s=2) = 4 independent 1x1 GEMMs with strided stores.
// in: [nImg][h][w][Cin]; wp: [4][Cout][Cin]; out: [nImg][2h][2w][Cout] (+bias)
// hw%16==0, Cout%16==0, Cin%32==0 at all call sites -> fully unguarded.
// ---------------------------------------------------------------------------
__global__ void __launch_bounds__(128)
k_convt2(const f16* __restrict__ in, const f16* __restrict__ wp,
         const float* __restrict__ bias, f16* __restrict__ out,
         int nImg, int h, int w, int Cin, int Cout, int tiles) {
  __shared__ __align__(16) f16 lds[4][2][512];
  const int lane = threadIdx.x & 31, wave = threadIdx.x >> 5;
  const int tile = blockIdx.x * 4 + wave;
  if (tile >= tiles) return;
  const int hw = h * w;
  const int mTiles = hw >> 4;
  const int nTiles = Cout >> 4;
  const int ab = tile & 3;
  int t1 = tile >> 2;
  const int nt = t1 % nTiles; t1 /= nTiles;
  const int mt = t1 % mTiles;
  const int img = t1 / mTiles;
  const int mBase = mt << 4, nBase = nt << 4;
  f16* As = lds[wave][0];
  f16* Bs = lds[wave][1];
  const int r = lane & 15, seg = lane >> 4;
  const f16* src  = in + ((long)img * hw + mBase + r) * Cin + (seg << 4);
  const f16* wrow = wp + ((long)ab * Cout + nBase + r) * Cin + (seg << 4);
  int4* da = (int4*)(As + r * 32 + (seg << 4));
  int4* db = (int4*)(Bs + r * 32 + (seg << 4));
  v8f acc = {};
  for (int k0 = 0; k0 < Cin; k0 += 32) {
    if (k0 + 32 < Cin) __builtin_prefetch(src + 32, 0, 3);
    da[0] = *(const int4*)src;
    da[1] = *(const int4*)(src + 8);
    db[0] = *(const int4*)wrow;
    db[1] = *(const int4*)(wrow + 8);
    __builtin_amdgcn_wave_barrier();
    wmma_acc_s(acc, As, Bs, lane, 32);
    __builtin_amdgcn_wave_barrier();
    src += 32;
    wrow += 32;
  }
  const int n = nBase + (lane & 15);
  const int hl = lane >> 4;
  const float bv = bias[n];
  #pragma unroll
  for (int j = 0; j < 8; ++j) {
    const int m = mBase + j + (hl << 3);
    const int yy = m / w, xx = m % w;
    const int oy = 2 * yy + (ab >> 1), ox = 2 * xx + (ab & 1);
    out[(((long)img * 2 * h + oy) * (2 * w) + ox) * Cout + n] = (f16)(acc[j] + bv);
  }
}

// ------------------------- pointwise / small kernels -----------------------

__global__ void k_pack_conv(const float* __restrict__ W, f16* __restrict__ wp,
                            int Cout, int Cin, int n) {
  long id = (long)blockIdx.x * blockDim.x + threadIdx.x;
  if (id >= n) return;
  const int tap = id % 9;
  long rest = id / 9;
  const int ci = rest % Cin;
  const int co = rest / Cin;
  wp[((long)tap * Cout + co) * Cin + ci] = (f16)W[((long)co * Cin + ci) * 9 + tap];
}

__global__ void k_pack_convt(const float* __restrict__ W, f16* __restrict__ wp,
                             int Cout, int Cin, int n) {
  long id = (long)blockIdx.x * blockDim.x + threadIdx.x;
  if (id >= n) return;
  const int ab = id & 3;
  long rest = id >> 2;
  const int co = rest % Cout;
  const int ci = rest / Cout;
  wp[((long)ab * Cout + co) * Cin + ci] = (f16)W[((long)ci * Cout + co) * 4 + ab];
}

__global__ void k_cast_f16(const float* __restrict__ src, f16* __restrict__ dst, long n) {
  long id = (long)blockIdx.x * blockDim.x + threadIdx.x;
  if (id < n) dst[id] = (f16)src[id];
}

__global__ void k_zero_state(float* __restrict__ c, f16* __restrict__ h, long n) {
  long id = (long)blockIdx.x * blockDim.x + threadIdx.x;
  if (id < n) { c[id] = 0.f; h[id] = (f16)0.f; }
}

// channel attention logits: one block per (b,c); reduce tanh(x)*W_ca
__global__ void k_ca_logits(const float* __restrict__ x, const float* __restrict__ Wca,
                            const float* __restrict__ bca, float* __restrict__ logits) {
  __shared__ float red[256];
  const int bc = blockIdx.x;
  const int b = bc >> 3, c = bc & 7;
  const float* xb = x + ((long)b * Tn * Cch + c) * 4096;  // + t*C*4096 + s
  float s = 0.f;
  for (int i = threadIdx.x; i < Tn * 4096; i += 256) {
    const int t = i >> 12, sp = i & 4095;
    s += tanhf(xb[(long)t * Cch * 4096 + sp]) * Wca[i];
  }
  red[threadIdx.x] = s;
  __syncthreads();
  for (int st = 128; st > 0; st >>= 1) {
    if ((int)threadIdx.x < st) red[threadIdx.x] += red[threadIdx.x + st];
    __syncthreads();
  }
  if (threadIdx.x == 0) logits[bc] = red[0] + bca[0];
}

__global__ void k_ca_softmax(const float* __restrict__ logits,
                             float* __restrict__ attn, float* __restrict__ out_attn) {
  const int b = threadIdx.x;
  if (b >= Bsz) return;
  float m = -1e30f;
  for (int c = 0; c < Cch; ++c) m = fmaxf(m, logits[b * Cch + c]);
  float den = 0.f, e[Cch];
  for (int c = 0; c < Cch; ++c) { e[c] = __expf(logits[b * Cch + c] - m); den += e[c]; }
  for (int c = 0; c < Cch; ++c) {
    const float v = e[c] / den;
    attn[b * Cch + c] = v;
    out_attn[b * Cch + c] = v;
  }
}

// scale + NCHW -> channel-last f16:  xw[(b*T+t)][s][c]
__global__ void k_scale(const float* __restrict__ x, const float* __restrict__ attn,
                        f16* __restrict__ xw, long n) {
  long id = (long)blockIdx.x * blockDim.x + threadIdx.x;
  if (id >= n) return;
  const long s = id & 4095;
  long r = id >> 12;
  const int c = r & 7;
  const long bt = r >> 3;
  const int b = (int)(bt / Tn);
  xw[(bt * 4096 + s) * Cch + c] = (f16)(x[id] * attn[b * Cch + c]);
}

__global__ void k_maxpool(const f16* __restrict__ in, f16* __restrict__ out,
                          int H, int C, long n) {
  long id = (long)blockIdx.x * blockDim.x + threadIdx.x;
  if (id >= n) return;
  const int c = id % C;
  long r = id / C;
  const int Wh = H >> 1;
  const int ox = r % Wh; r /= Wh;
  const int oy = r % Wh;
  const long img = r / Wh;
  const f16* p = in + (((long)img * H + 2 * oy) * H + 2 * ox) * C + c;
  float a = (float)p[0];
  a = fmaxf(a, (float)p[C]);
  a = fmaxf(a, (float)p[(long)H * C]);
  a = fmaxf(a, (float)p[(long)H * C + C]);
  out[id] = (f16)a;
}

// c4 [(b*T+t)][pix][128] -> feat [(b*64+pix)*T + t][128]
__global__ void k_featperm(const f16* __restrict__ c4, f16* __restrict__ feat, long n) {
  long id = (long)blockIdx.x * blockDim.x + threadIdx.x;
  if (id >= n) return;
  const int c = id & 127;
  long r = id >> 7;
  const int t = r % Tn;
  const long nf = r / Tn;              // (b,pix)
  const int b = (int)(nf >> 6), pix = (int)(nf & 63);
  feat[id] = c4[(((long)(b * Tn + t)) * 64 + pix) * 128 + c];
}

// LSTM cell update (128-wide hidden): z = Zin[n*T+t] + Hterm[n]
__global__ void k_lstm_step(const float* __restrict__ Zin, const float* __restrict__ Hterm,
                            float* __restrict__ cst, f16* __restrict__ hst,
                            f16* __restrict__ lstm_out, int t, int off) {
  long id = (long)blockIdx.x * blockDim.x + threadIdx.x;
  if (id >= 1024L * 128) return;
  const long n = id >> 7;
  const int j = id & 127;
  const float* z0 = Zin + (n * Tn + t) * 512;
  const float* hT = Hterm + n * 512;
  const float zi = z0[j] + hT[j];
  const float zf = z0[128 + j] + hT[128 + j];
  const float zg = z0[256 + j] + hT[256 + j];
  const float zo = z0[384 + j] + hT[384 + j];
  const float c = sigf(zf) * cst[id] + sigf(zi) * tanhf(zg);
  const float hh = sigf(zo) * tanhf(c);
  cst[id] = c;
  hst[id] = (f16)hh;
  lstm_out[(n * Tn + t) * 256 + off + j] = (f16)(hh > 0.f ? hh : 0.f);
}

// attention scores: one wave per (n,t), dot(tanh(lstm_row), W_att)
__global__ void __launch_bounds__(128)
k_scores(const f16* __restrict__ lstm, const float* __restrict__ Watt,
         const float* __restrict__ batt, float* __restrict__ scores, int items) {
  const int item = blockIdx.x * 4 + (threadIdx.x >> 5);
  if (item >= items) return;
  const int lane = threadIdx.x & 31;
  const f16* row = lstm + (long)item * 256;
  float s = 0.f;
  for (int d = lane; d < 256; d += 32) s += tanhf((float)row[d]) * Watt[d];
  s = wave_reduce_sum(s);
  if (lane == 0) scores[item] = s + batt[0];
}

__global__ void k_aw(const float* __restrict__ scores, float* __restrict__ aw_ws,
                     float* __restrict__ out_aw) {
  __shared__ float sv[Tn];
  __shared__ float red[2];
  const int b = blockIdx.x;
  const int t = threadIdx.x;
  if (t < Tn) {
    float s = 0.f;
    for (int p = 0; p < 64; ++p) s += scores[((long)(b * 64 + p)) * Tn + t];
    sv[t] = s * (1.f / 64.f);
  }
  __syncthreads();
  if (t == 0) {
    float m = -1e30f;
    for (int i = 0; i < Tn; ++i) m = fmaxf(m, sv[i]);
    float d = 0.f;
    for (int i = 0; i < Tn; ++i) d += __expf(sv[i] - m);
    red[0] = m; red[1] = d;
  }
  __syncthreads();
  if (t < Tn) {
    const float v = __expf(sv[t] - red[0]) / red[1];
    aw_ws[b * Tn + t] = v;
    out_aw[b * Tn + t] = v;
  }
}

// ctx[n][d] = sum_t aw[b][t]*lstm[n][t][d]; also seeds decoder state.
__global__ void k_ctx(const f16* __restrict__ lstm, const float* __restrict__ aw_ws,
                      float* __restrict__ ctxf, f16* __restrict__ inp,
                      f16* __restrict__ hdec, float* __restrict__ cdec) {
  long id = (long)blockIdx.x * blockDim.x + threadIdx.x;
  if (id >= 1024L * 256) return;
  const int d = id & 255;
  const long n = id >> 8;
  const int b = (int)(n >> 6);
  const f16* base = lstm + n * Tn * 256 + d;
  float s = 0.f;
  for (int t = 0; t < Tn; ++t) s += aw_ws[b * Tn + t] * (float)base[(long)t * 256];
  ctxf[id] = s;
  inp[id] = (f16)s;
  hdec[id] = (f16)0.f;
  cdec[id] = 0.f;
}

__global__ void k_codevec(const float* __restrict__ ctxf, float* __restrict__ outv) {
  long id = (long)blockIdx.x * blockDim.x + threadIdx.x;
  if (id >= Bsz * 256) return;
  const int d = id & 255, b = (int)(id >> 8);
  float s = 0.f;
  for (int p = 0; p < 64; ++p) s += ctxf[((long)(b * 64 + p)) * 256 + d];
  outv[b * 256 + d] = s * (1.f / 64.f);
}

// decoder LSTM cell (256-wide): z = G1 + G2 + bias_d; writes dec image layout.
__global__ void k_dec_step(const float* __restrict__ G1, const float* __restrict__ G2,
                           const float* __restrict__ bias_d, float* __restrict__ cdec,
                           f16* __restrict__ hdec, f16* __restrict__ inp,
                           const float* __restrict__ ctxf, f16* __restrict__ dec_l, int t) {
  long id = (long)blockIdx.x * blockDim.x + threadIdx.x;
  if (id >= 1024L * 256) return;
  const long n = id >> 8;
  const int j = id & 255;
  const float* g1 = G1 + n * 1024;
  const float* g2 = G2 + n * 1024;
  const float zi = g1[j] + g2[j] + bias_d[j];
  const float zf = g1[256 + j] + g2[256 + j] + bias_d[256 + j];
  const float zg = g1[512 + j] + g2[512 + j] + bias_d[512 + j];
  const float zo = g1[768 + j] + g2[768 + j] + bias_d[768 + j];
  const float c = sigf(zf) * cdec[id] + sigf(zi) * tanhf(zg);
  const float hh = sigf(zo) * tanhf(c);
  cdec[id] = c;
  hdec[id] = (f16)hh;
  inp[id] = (f16)(hh + ctxf[id]);
  const int b = (int)(n >> 6), pix = (int)(n & 63);
  dec_l[(((long)(b * Tn + t)) * 64 + pix) * 256 + j] = (f16)hh;
}

// final 1x1 conv 16->8, fp32 store into d_out image section ([B,T,C,P,P])
__global__ void k_outconv(const f16* __restrict__ u1, const float* __restrict__ Wout,
                          const float* __restrict__ bout, float* __restrict__ outp, long n) {
  long id = (long)blockIdx.x * blockDim.x + threadIdx.x;
  if (id >= n) return;
  const int c = id & 7;
  const long s = (id >> 3) & 4095;
  const long img = id >> 15;
  const f16* row = u1 + (img * 4096 + s) * 16;
  const float* wo = Wout + c * 16;
  float v = bout[c];
  #pragma unroll
  for (int k = 0; k < 16; ++k) v += (float)row[k] * wo[k];
  outp[OUT_IMG + (img * Cch + c) * 4096 + s] = v;
}

// ---------------------------------------------------------------------------
extern "C" void kernel_launch(void* const* d_in, const int* in_sizes, int n_in,
                              void* d_out, int out_size, void* d_ws, size_t ws_size,
                              hipStream_t stream) {
  (void)in_sizes; (void)n_in; (void)out_size; (void)ws_size;
  const float* x = (const float*)d_in[0];
  auto P = [&](int i) { return (const float*)d_in[i]; };
  // input index map (dict insertion order of setup_inputs)
  // 1..22: (W,b) for c11,c12,c21,c22,c31,c32,c41,c42,u32,u22,u12
  // 23/24: W_outc/b_outc; 25..30: (W,b) t3,t2,t1
  // 31..36: Wih_f,Whh_f,bias_f,Wih_b,Whh_b,bias_b; 37..39: Wih_d,Whh_d,bias_d
  // 40/41: W_att/b_att; 42/43: W_ca/b_ca
  float* outp = (float*)d_out;

  char* base = (char*)d_ws;
  size_t off = 0;
  auto alloc = [&](size_t bytes) -> void* {
    void* p = base + off;
    off = (off + bytes + 255) & ~(size_t)255;
    return p;
  };

  f16* xw   = (f16*)alloc((size_t)NIMG * 4096 * 8 * 2);
  f16* actA = (f16*)alloc((size_t)NIMG * 4096 * 16 * 2);
  f16* actB = (f16*)alloc((size_t)NIMG * 4096 * 16 * 2);
  // packed conv weights
  f16* wpc[11];
  const int convCi[11] = {8, 16, 16, 32, 32, 64, 64, 128, 64, 32, 16};
  const int convCo[11] = {16, 16, 32, 32, 64, 64, 128, 128, 64, 32, 16};
  for (int i = 0; i < 11; ++i) wpc[i] = (f16*)alloc((size_t)9 * convCo[i] * convCi[i] * 2);
  f16* wpt3 = (f16*)alloc((size_t)4 * 64 * 256 * 2);
  f16* wpt2 = (f16*)alloc((size_t)4 * 32 * 64 * 2);
  f16* wpt1 = (f16*)alloc((size_t)4 * 16 * 32 * 2);
  f16* Wihf = (f16*)alloc((size_t)512 * 128 * 2);
  f16* Whhf = (f16*)alloc((size_t)512 * 128 * 2);
  f16* Wihb = (f16*)alloc((size_t)512 * 128 * 2);
  f16* Whhb = (f16*)alloc((size_t)512 * 128 * 2);
  f16* Wihd = (f16*)alloc((size_t)1024 * 256 * 2);
  f16* Whhd = (f16*)alloc((size_t)1024 * 256 * 2);
  float* logits = (float*)alloc(128 * 4);
  float* attnca = (float*)alloc(128 * 4);
  f16*  feat    = (f16*)alloc((size_t)24576 * 128 * 2);
  float* ZinF   = (float*)alloc((size_t)24576 * 512 * 4);
  float* ZinB   = (float*)alloc((size_t)24576 * 512 * 4);
  float* Hterm  = (float*)alloc((size_t)1024 * 512 * 4);
  float* cst    = (float*)alloc((size_t)1024 * 128 * 4);
  f16*   hst    = (f16*)alloc((size_t)1024 * 128 * 2);
  f16*   lstm   = (f16*)alloc((size_t)1024 * Tn * 256 * 2);
  float* scores = (float*)alloc((size_t)1024 * Tn * 4);
  float* aw_ws  = (float*)alloc((size_t)Bsz * Tn * 4);
  float* ctxf   = (float*)alloc((size_t)1024 * 256 * 4);
  f16*   dinp   = (f16*)alloc((size_t)1024 * 256 * 2);
  f16*   hdec   = (f16*)alloc((size_t)1024 * 256 * 2);
  float* cdec   = (float*)alloc((size_t)1024 * 256 * 4);
  float* G1     = (float*)alloc((size_t)1024 * 1024 * 4);
  float* G2     = (float*)alloc((size_t)1024 * 1024 * 4);
  f16*   dec_l  = (f16*)alloc((size_t)NIMG * 64 * 256 * 2);

  auto grid1 = [](long n) { return (unsigned)((n + 255) / 256); };

  // ---- weight prep ----
  const int convWidx[11] = {1, 3, 5, 7, 9, 11, 13, 15, 17, 19, 21};
  for (int i = 0; i < 11; ++i) {
    const int n = convCo[i] * convCi[i] * 9;
    k_pack_conv<<<grid1(n), 256, 0, stream>>>(P(convWidx[i]), wpc[i], convCo[i], convCi[i], n);
  }
  k_pack_convt<<<grid1(256 * 64 * 4), 256, 0, stream>>>(P(25), wpt3, 64, 256, 256 * 64 * 4);
  k_pack_convt<<<grid1(64 * 32 * 4), 256, 0, stream>>>(P(27), wpt2, 32, 64, 64 * 32 * 4);
  k_pack_convt<<<grid1(32 * 16 * 4), 256, 0, stream>>>(P(29), wpt1, 16, 32, 32 * 16 * 4);
  k_cast_f16<<<grid1(512 * 128), 256, 0, stream>>>(P(31), Wihf, 512 * 128);
  k_cast_f16<<<grid1(512 * 128), 256, 0, stream>>>(P(32), Whhf, 512 * 128);
  k_cast_f16<<<grid1(512 * 128), 256, 0, stream>>>(P(34), Wihb, 512 * 128);
  k_cast_f16<<<grid1(512 * 128), 256, 0, stream>>>(P(35), Whhb, 512 * 128);
  k_cast_f16<<<grid1(1024 * 256), 256, 0, stream>>>(P(37), Wihd, 1024 * 256);
  k_cast_f16<<<grid1(1024 * 256), 256, 0, stream>>>(P(38), Whhd, 1024 * 256);

  // ---- channel attention ----
  k_ca_logits<<<128, 256, 0, stream>>>(x, P(42), P(43), logits);
  k_ca_softmax<<<1, 32, 0, stream>>>(logits, attnca, outp + OUT_CA);
  k_scale<<<grid1((long)NIMG * Cch * 4096), 256, 0, stream>>>(x, attnca, xw,
                                                              (long)NIMG * Cch * 4096);

  // ---- UNet encoder ----
  auto conv = [&](const f16* in, int li, const float* bias, f16* out, int H, int relu) {
    const int HW = H * H;
    const int tiles = NIMG * (HW / 16) * (convCo[li] / 16);
    k_conv3x3<<<(tiles + 3) / 4, 128, 0, stream>>>(in, wpc[li], bias, out, NIMG, H, H,
                                                   convCi[li], convCo[li], relu, tiles);
  };
  auto pool = [&](const f16* in, f16* out, int H, int C) {
    const long n = (long)NIMG * (H / 2) * (H / 2) * C;
    k_maxpool<<<grid1(n), 256, 0, stream>>>(in, out, H, C, n);
  };
  conv(xw,   0, P(2),  actA, 64, 1);   // c11
  conv(actA, 1, P(4),  actB, 64, 1);   // c12
  pool(actB, actA, 64, 16);
  conv(actA, 2, P(6),  actB, 32, 1);   // c21
  conv(actB, 3, P(8),  actA, 32, 1);   // c22
  pool(actA, actB, 32, 32);
  conv(actB, 4, P(10), actA, 16, 1);   // c31
  conv(actA, 5, P(12), actB, 16, 1);   // c32
  pool(actB, actA, 16, 64);
  conv(actA, 6, P(14), actB, 8, 1);    // c41
  conv(actB, 7, P(16), actA, 8, 1);    // c42 -> c4 in actA

  // ---- BiLSTM ----
  k_featperm<<<grid1(24576L * 128), 256, 0, stream>>>(actA, feat, 24576L * 128);
  auto gemm = [&](const f16* A, const f16* Bw, const float* bias, float* Cf, f16* Ch,
                  int M, int N, int K, int relu) {
    const int tiles = (M / 16) * (N / 16);
    k_gemm_nt<<<(tiles + 3) / 4, 128, 0, stream>>>(A, Bw, bias, Cf, Ch, M, N, K, relu, tiles);
  };
  gemm(feat, Wihf, P(33), ZinF, nullptr, 24576, 512, 128, 0);
  gemm(feat, Wihb, P(36), ZinB, nullptr, 24576, 512, 128, 0);

  k_zero_state<<<grid1(1024L * 128), 256, 0, stream>>>(cst, hst, 1024L * 128);
  for (int t = 0; t < Tn; ++t) {
    gemm(hst, Whhf, nullptr, Hterm, nullptr, 1024, 512, 128, 0);
    k_lstm_step<<<grid1(1024L * 128), 256, 0, stream>>>(ZinF, Hterm, cst, hst, lstm, t, 0);
  }
  k_zero_state<<<grid1(1024L * 128), 256, 0, stream>>>(cst, hst, 1024L * 128);
  for (int s = 0; s < Tn; ++s) {
    const int t = Tn - 1 - s;
    gemm(hst, Whhb, nullptr, Hterm, nullptr, 1024, 512, 128, 0);
    k_lstm_step<<<grid1(1024L * 128), 256, 0, stream>>>(ZinB, Hterm, cst, hst, lstm, t, 128);
  }

  // ---- temporal attention + context ----
  k_scores<<<(24576 + 3) / 4, 128, 0, stream>>>(lstm, P(40), P(41), scores, 24576);
  k_aw<<<Bsz, 32, 0, stream>>>(scores, aw_ws, outp + OUT_AW);
  k_ctx<<<grid1(1024L * 256), 256, 0, stream>>>(lstm, aw_ws, ctxf, dinp, hdec, cdec);
  k_codevec<<<grid1(Bsz * 256), 256, 0, stream>>>(ctxf, outp + OUT_CODE);

  // ---- decoder LSTM ----
  for (int t = 0; t < Tn; ++t) {
    gemm(dinp, Wihd, nullptr, G1, nullptr, 1024, 1024, 256, 0);
    gemm(hdec, Whhd, nullptr, G2, nullptr, 1024, 1024, 256, 0);
    k_dec_step<<<grid1(1024L * 256), 256, 0, stream>>>(G1, G2, P(39), cdec, hdec, dinp,
                                                       ctxf, dec_l, t);
  }

  // ---- UNet decoder ----
  auto convt = [&](const f16* in, const f16* wp, const float* bias, f16* out,
                   int h, int Cin, int Cout) {
    const int tiles = NIMG * ((h * h) / 16) * (Cout / 16) * 4;
    k_convt2<<<(tiles + 3) / 4, 128, 0, stream>>>(in, wp, bias, out, NIMG, h, h, Cin, Cout, tiles);
  };
  convt(dec_l, wpt3, P(26), actA, 8, 256, 64);   // t3 -> 16x16x64
  conv(actA, 8, P(18), actB, 16, 1);             // u32
  convt(actB, wpt2, P(28), actA, 16, 64, 32);    // t2 -> 32x32x32
  conv(actA, 9, P(20), actB, 32, 1);             // u22
  convt(actB, wpt1, P(30), actA, 32, 32, 16);    // t1 -> 64x64x16
  conv(actA, 10, P(22), actB, 64, 1);            // u12

  // ---- out conv (1x1, fp32) ----
  const long nOut = (long)NIMG * 4096 * 8;
  k_outconv<<<grid1(nOut), 256, 0, stream>>>(actB, P(23), P(24), outp, nOut);
}
